// AR_56822417326097
// MI455X (gfx1250) — matrix operand order repaired and checked by reading, not compiled, not run on hardware
//
#include <hip/hip_runtime.h>
#include <math.h>

#define D_   32
#define H_   16
#define L_   8
#define MLP_ 128
#define B_   16
#define S_   1024
#define T_   256
#define P_   32
#define BS_  (B_ * S_)

typedef __attribute__((ext_vector_type(16))) __bf16 v16bf;
typedef __attribute__((ext_vector_type(8)))  float  v8f;

#define GF_ACCUM 1
#define GF_GELU  2

// ---------------------------------------------------------------------------
// bf16 WMMA GEMM:  C(M,N) [= or +=]  act( A(M,K) @ W(N,K)^T + bias(N) )
// K = NK*32 compile-time. One wave owns a 16-row strip of A: it preloads all
// NK A-fragments into registers (float4 global loads), then sweeps every
// 16-col tile of N reusing them -> one v_wmma_f32_16x16x32_bf16 per (tile,ks).
// M, N multiples of 16.
// ---------------------------------------------------------------------------
template <int NK>
__global__ __launch_bounds__(128) void gemm_wmma_kernel(
    const float* __restrict__ A, const float* __restrict__ W,
    const float* __restrict__ bias, float* __restrict__ C,
    int M, int N, int ldc, int flags)
{
    constexpr int K = NK * 32;
    int lane = threadIdx.x & 31;
    int mt   = blockIdx.x * (blockDim.x >> 5) + (threadIdx.x >> 5);
    if (mt >= (M >> 4)) return;                 // wave-uniform: EXEC stays all-1s
    int tm = mt << 4;

    // ISA 7.12.2 fragment layouts (wave32, 16-bit operands)
    int ka = (lane & 16) ? 8  : 0;              // A lane-half K offset
    int kb = (lane & 16) ? 8  : 0;              // B lane-half K offset
    const float* Arow = A + (size_t)(tm + (lane & 15)) * K;

    v16bf afrag[NK];
#pragma unroll
    for (int ks = 0; ks < NK; ++ks) {
        const float* ap = Arow + ks * 32 + ka;
        float4 l0 = *(const float4*)(ap + 0);   // K = ka+0..3
        float4 l1 = *(const float4*)(ap + 4);   // K = ka+4..7
        float4 h0 = *(const float4*)(ap + 16);  // K = ka+16..19
        float4 h1 = *(const float4*)(ap + 20);  // K = ka+20..23
        float tmp[16] = {l0.x, l0.y, l0.z, l0.w, l1.x, l1.y, l1.z, l1.w,
                         h0.x, h0.y, h0.z, h0.w, h1.x, h1.y, h1.z, h1.w};
#pragma unroll
        for (int e = 0; e < 16; ++e) afrag[ks][e] = (__bf16)tmp[e];
    }

    int tilesN = N >> 4;
    for (int tn = 0; tn < tilesN; ++tn) {
        const float* Wrow = W + (size_t)((tn << 4) + (lane & 15)) * K + kb;
        v8f acc = {0.f, 0.f, 0.f, 0.f, 0.f, 0.f, 0.f, 0.f};
#pragma unroll
        for (int ks = 0; ks < NK; ++ks) {
            const float* wp = Wrow + ks * 32;
            float4 b0 = *(const float4*)(wp + 0);
            float4 b1 = *(const float4*)(wp + 4);
            float4 b2 = *(const float4*)(wp + 16);
            float4 b3 = *(const float4*)(wp + 20);
            float tb[16] = {b0.x, b0.y, b0.z, b0.w, b1.x, b1.y, b1.z, b1.w,
                            b2.x, b2.y, b2.z, b2.w, b3.x, b3.y, b3.z, b3.w};
            v16bf bfrag;
#pragma unroll
            for (int e = 0; e < 16; ++e) bfrag[e] = (__bf16)tb[e];
            acc = __builtin_amdgcn_wmma_f32_16x16x32_bf16(
                      false, afrag[ks], false, bfrag, (short)0, acc, false, false);
        }

        int   cm   = tm + ((lane & 16) ? 8 : 0);    // C: VGPR r -> M = cm + r
        int   cn   = (tn << 4) + (lane & 15);
        float bval = bias ? bias[cn] : 0.f;
#pragma unroll
        for (int r = 0; r < 8; ++r) {
            float v = acc[r] + bval;
            if (flags & GF_GELU) v = 0.5f * v * (1.f + erff(v * 0.70710678118654752f));
            float* cp = C + (size_t)(cm + r) * ldc + cn;
            if (flags & GF_ACCUM) *cp += v; else *cp = v;
        }
    }
}

// ---------------------------------------------------------------------------
// LayerNorm over D=32 (one wave per token). Optional: in-place h *= mask first,
// optional timestep scale/shift modulation (ss is (B,64): [:,0:32]=scale,
// [:,32:64]=shift;  y = ln(x)*(1+scale) + shift).
// ---------------------------------------------------------------------------
__global__ void ln_kernel(float* __restrict__ h, const float* __restrict__ g,
                          const float* __restrict__ b, float* __restrict__ out,
                          const float* __restrict__ mask,
                          const float* __restrict__ ss, int ntok)
{
    int wid  = (blockIdx.x * blockDim.x + threadIdx.x) >> 5;
    int lane = threadIdx.x & 31;
    if (wid >= ntok) return;
    size_t idx = (size_t)wid * 32 + lane;
    float x = h[idx];
    if (mask) { x *= mask[wid & (S_ - 1)]; h[idx] = x; }

    float s = x;
#pragma unroll
    for (int o = 16; o > 0; o >>= 1) s += __shfl_xor(s, o, 32);
    float mu = s * (1.f / 32.f);
    float d  = x - mu;
    float vv = d * d;
#pragma unroll
    for (int o = 16; o > 0; o >>= 1) vv += __shfl_xor(vv, o, 32);
    float y = d * rsqrtf(vv * (1.f / 32.f) + 1e-6f) * g[lane] + b[lane];
    if (ss) {
        int bt = wid >> 10;                     // token / S
        y = y * (1.f + ss[bt * 64 + lane]) + ss[bt * 64 + 32 + lane];
    }
    out[idx] = y;
}

// ---------------------------------------------------------------------------
// Online-softmax attention, head_dim = 2. Block = 256 threads = 16 queries x
// 16 heads of one batch. K/V rows (64 contiguous floats: k[0:32] v[32:64]
// starting at column `kvoff`) are staged per 64-key chunk into LDS with
// coalesced float4 copies; inner loop streams from LDS. Next chunk prefetched.
// ---------------------------------------------------------------------------
#define CH_ 64
__global__ __launch_bounds__(256) void attn_kernel(
    const float* __restrict__ q, int ldq,
    const float* __restrict__ kv, int ldkv, int kvoff,
    float* __restrict__ out, int Sq, int Sk)
{
    __shared__ float tile[CH_ * 64];            // 16 KB of the 320 KB WGP LDS
    int tid = threadIdx.x;
    int hh  = tid & 15;
    int qr  = tid >> 4;
    int bb  = blockIdx.y;
    int qi  = (blockIdx.x << 4) + qr;

    const float* qp = q + ((size_t)bb * Sq + qi) * ldq + hh * 2;
    const float  sc = 0.70710678118654752f;     // 1/sqrt(head_dim=2)
    float q0 = qp[0] * sc, q1 = qp[1] * sc;

    const float* kvb = kv + (size_t)bb * Sk * ldkv + kvoff;
    float m = -3.4e38f, l = 0.f, a0 = 0.f, a1 = 0.f;

    for (int j0 = 0; j0 < Sk; j0 += CH_) {
        int rem = Sk - j0; if (rem > CH_) rem = CH_;
        int nf4 = rem << 4;                     // float4 transfers this chunk
        __syncthreads();                        // previous chunk fully consumed
        for (int f = tid; f < nf4; f += 256) {
            int row = f >> 4, c4 = (f & 15) << 2;
            *(float4*)(tile + row * 64 + c4) =
                *(const float4*)(kvb + (size_t)(j0 + row) * ldkv + c4);
        }
        if (j0 + CH_ < Sk)                      // overlap next chunk (global_prefetch)
            __builtin_prefetch(kvb + (size_t)(j0 + CH_ + (tid >> 2)) * ldkv, 0, 0);
        __syncthreads();

        for (int jj = 0; jj < rem; ++jj) {
            const float* tr = tile + jj * 64;
            float2 kk = *(const float2*)(tr + hh * 2);
            float2 vv = *(const float2*)(tr + 32 + hh * 2);
            float s    = q0 * kk.x + q1 * kk.y;
            float mn   = fmaxf(m, s);
            float corr = __expf(m - mn);
            float p    = __expf(s - mn);
            l  = l  * corr + p;
            a0 = a0 * corr + p * vv.x;
            a1 = a1 * corr + p * vv.y;
            m  = mn;
        }
    }
    float inv = 1.f / l;
    float* op = out + ((size_t)bb * Sq + qi) * 32 + hh * 2;
    op[0] = a0 * inv;
    op[1] = a1 * inv;
}

__global__ void mask_kernel(float* __restrict__ h, const float* __restrict__ mask, int n)
{
    int t = blockIdx.x * blockDim.x + threadIdx.x;
    if (t >= n) return;
    h[t] *= mask[(t >> 5) & (S_ - 1)];
}

// ---------------------------------------------------------------------------
extern "C" void kernel_launch(void* const* d_in, const int* in_sizes, int n_in,
                              void* d_out, int out_size, void* d_ws, size_t ws_size,
                              hipStream_t stream)
{
    const float* x      = (const float*)d_in[0];
    const float* spk    = (const float*)d_in[1];
    const float* text   = (const float*)d_in[2];
    const float* tse    = (const float*)d_in[3];
    const float* mask   = (const float*)d_in[4];
    const float* ln1_g  = (const float*)d_in[5];
    const float* ln1_b  = (const float*)d_in[6];
    const float* ln2_g  = (const float*)d_in[7];
    const float* ln2_b  = (const float*)d_in[8];
    const float* ln22_g = (const float*)d_in[9];
    const float* ln22_b = (const float*)d_in[10];
    const float* ln3_g  = (const float*)d_in[11];
    const float* ln3_b  = (const float*)d_in[12];
    const float* aw_in  = (const float*)d_in[13];
    const float* ab_in  = (const float*)d_in[14];
    const float* aw_out = (const float*)d_in[15];
    const float* ab_out = (const float*)d_in[16];
    const float* cw_in  = (const float*)d_in[17];
    const float* cb_in  = (const float*)d_in[18];
    const float* cw_out = (const float*)d_in[19];
    const float* cb_out = (const float*)d_in[20];
    const float* ts_w   = (const float*)d_in[21];
    const float* ts_b   = (const float*)d_in[22];
    const float* f1w    = (const float*)d_in[23];
    const float* f1b    = (const float*)d_in[24];
    const float* f2w    = (const float*)d_in[25];
    const float* f2b    = (const float*)d_in[26];

    float* h = (float*)d_out;                   // residual stream lives in d_out
    float* p = (float*)d_ws;
    float* hn  = p; p += (size_t)BS_ * 32;
    float* qkv = p; p += (size_t)BS_ * 96;
    float* ao  = p; p += (size_t)BS_ * 32;
    float* qc1 = p; p += (size_t)BS_ * 32;
    float* qc2 = p; p += (size_t)BS_ * 32;
    float* kvt = p; p += (size_t)B_ * T_ * 64;
    float* kvs = p; p += (size_t)B_ * P_ * 64;
    float* ssb = p; p += (size_t)B_ * 64;
    float* hm  = p; p += (size_t)BS_ * 32;
    float* t1  = p; p += (size_t)BS_ * 128;

    hipMemcpyAsync(h, x, (size_t)BS_ * 32 * sizeof(float),
                   hipMemcpyDeviceToDevice, stream);

    auto gemm32 = [&](const float* A, const float* W, const float* bias, float* C,
                      int M, int N, int ldc, int flags) {
        int blocks = ((M >> 4) + 3) / 4;
        gemm_wmma_kernel<1><<<blocks, 128, 0, stream>>>(A, W, bias, C, M, N, ldc, flags);
    };
    auto gemm128 = [&](const float* A, const float* W, const float* bias, float* C,
                       int M, int N, int ldc, int flags) {
        int blocks = ((M >> 4) + 3) / 4;
        gemm_wmma_kernel<4><<<blocks, 128, 0, stream>>>(A, W, bias, C, M, N, ldc, flags);
    };

    const int  lnBlocks = (BS_ * 32) / 256;
    const dim3 attnGrid(S_ / 16, B_);

    for (int l = 0; l < L_; ++l) {
        const float* wi  = aw_in  + (size_t)l * 96 * 32;
        const float* bi  = ab_in  + (size_t)l * 96;
        const float* wo  = aw_out + (size_t)l * 32 * 32;
        const float* bo  = ab_out + (size_t)l * 32;
        const float* cwi = cw_in  + (size_t)l * 96 * 32;
        const float* cbi = cb_in  + (size_t)l * 96;
        const float* cwo = cw_out + (size_t)l * 32 * 32;
        const float* cbo = cb_out + (size_t)l * 32;

        // h *= mask ; hn = LN1(h)
        ln_kernel<<<lnBlocks, 256, 0, stream>>>(h, ln1_g + l * 32, ln1_b + l * 32,
                                                hn, mask, nullptr, BS_);
        // qkv = hn @ in_w^T + in_b   (N=96: q|k|v)
        gemm32(hn, wi, bi, qkv, BS_, 96, 96, 0);
        // self-attention (Sk = S); k at col 32, v at col 64 -> kvoff = 32
        attn_kernel<<<attnGrid, 256, 0, stream>>>(qkv, 96, qkv, 96, 32, ao, S_, S_);
        // h += ao @ out_w^T + out_b
        gemm32(ao, wo, bo, h, BS_, 32, 32, GF_ACCUM);

        // both cross LNs read h BEFORE any cross output is accumulated
        ln_kernel<<<lnBlocks, 256, 0, stream>>>(h, ln2_g + l * 32, ln2_b + l * 32,
                                                hn, nullptr, nullptr, BS_);
        gemm32(hn, cwi, cbi, qc1, BS_, 32, 32, 0);
        ln_kernel<<<lnBlocks, 256, 0, stream>>>(h, ln22_g + l * 32, ln22_b + l * 32,
                                                hn, nullptr, nullptr, BS_);
        gemm32(hn, cwi, cbi, qc2, BS_, 32, 32, 0);

        // K,V from text / speaker:  rows 32..95 of cross_in_w
        gemm32(text, cwi + 32 * 32, cbi + 32, kvt, B_ * T_, 64, 64, 0);
        gemm32(spk,  cwi + 32 * 32, cbi + 32, kvs, B_ * P_, 64, 64, 0);

        // cross-attn 1 (text, Sk=T) then accumulate projection
        attn_kernel<<<attnGrid, 256, 0, stream>>>(qc1, 32, kvt, 64, 0, ao, S_, T_);
        gemm32(ao, cwo, cbo, h, BS_, 32, 32, GF_ACCUM);
        // cross-attn 2 (speaker, Sk=P)
        attn_kernel<<<attnGrid, 256, 0, stream>>>(qc2, 32, kvs, 64, 0, ao, S_, P_);
        gemm32(ao, cwo, cbo, h, BS_, 32, 32, GF_ACCUM);

        // timestep scale/shift:  ss = tse @ ts_w^T + ts_b   (M=16, N=64)
        gemm32(tse, ts_w + (size_t)l * 64 * 32, ts_b + l * 64, ssb, B_, 64, 64, 0);
        // hm = LN3(h)*(1+scale) + shift
        ln_kernel<<<lnBlocks, 256, 0, stream>>>(h, ln3_g + l * 32, ln3_b + l * 32,
                                                hm, nullptr, ssb, BS_);
        // t1 = gelu(hm @ fc1^T + b1) ;  h += t1 @ fc2^T + b2 ;  h *= mask
        gemm32 (hm, f1w + (size_t)l * 128 * 32, f1b + l * 128, t1, BS_, 128, 128, GF_GELU);
        gemm128(t1, f2w + (size_t)l * 32 * 128, f2b + l * 32,  h,  BS_, 32,  32,  GF_ACCUM);
        mask_kernel<<<lnBlocks, 256, 0, stream>>>(h, mask, BS_ * 32);
    }
}
